// _RumorGCN_66486093742679
// MI455X (gfx1250) — 16-bit and 8-bit matrix operand order repaired and checked by reading, compiled, not run on hardware
//
#include <hip/hip_runtime.h>
#include <hip/hip_bf16.h>


#define N_    16384
#define E_    16384
#define B_    64
#define IN_   768
#define H_    64
#define OUT_  64
#define EN_   4
#define HIN_  832
#define EPS_  1e-5f
#define EW    4

typedef __attribute__((ext_vector_type(16))) _Float16 v16h;
typedef __attribute__((ext_vector_type(8)))  _Float16 v8h;
typedef __attribute__((ext_vector_type(8)))  float    v8f;
typedef __attribute__((ext_vector_type(8)))  int      v8i;


__device__ __forceinline__ unsigned f32_to_fp8(float x) {
  union { float f; unsigned u; } v; v.f = x;
  unsigned s = v.u >> 31;
  int e = (int)((v.u >> 23) & 0xffu) - 127;
  unsigned m = (v.u >> 20) & 0x7u;
  int eb = e + 7;
  if (eb <= 0) return s << 7;
  if (eb > 15) { eb = 15; m = 6; }
  return (s << 7) | ((unsigned)eb << 3) | m;
}

__device__ __forceinline__ float sigmoidf_(float x) { return 1.f / (1.f + expf(-x)); }


__global__ void k_zero(float* __restrict__ p, int n) {
  int i = blockIdx.x * blockDim.x + threadIdx.x;
  if (i < n) p[i] = 0.f;
}

__global__ void k_f32_to_f16(const float* __restrict__ s, _Float16* __restrict__ d, int n) {
  int i = blockIdx.x * blockDim.x + threadIdx.x;
  if (i < n) d[i] = (_Float16)s[i];
}

__global__ void k_f32_to_fp8x4(const float* __restrict__ s, unsigned* __restrict__ d, int nwords) {
  int i = blockIdx.x * blockDim.x + threadIdx.x;
  if (i < nwords) {
    unsigned p = 0;
#pragma unroll
    for (int t = 0; t < 4; ++t) p |= f32_to_fp8(s[i * 4 + t]) << (8 * t);
    d[i] = p;
  }
}


__global__ __launch_bounds__(32) void k_wmma_gemm_f16(
    const _Float16* __restrict__ A, const _Float16* __restrict__ W,
    float* __restrict__ C, int M, int Nc, int K)
{
  const int ntiles = Nc >> 4;
  const int mo = blockIdx.x / ntiles;
  const int co = blockIdx.x - mo * ntiles;
  const int lane = threadIdx.x;
  const int rl = lane & 15, hi = lane >> 4;

  const _Float16* pa = A + (size_t)(mo * 16 + rl) * K + hi * 16;
  const _Float16* pb = W + (size_t)(co * 16 + rl) * K + hi * 16;
  v8f acc = {};
  for (int kb = 0; kb < K; kb += 32) {
    __builtin_prefetch(pa + kb + 256, 0, 0);
    v8h a0 = *(const v8h*)(pa + kb);
    v8h a1 = *(const v8h*)(pa + kb + 8);
    v16h a, b;
    b = *(const v16h*)(pb + kb);
#pragma unroll
    for (int i = 0; i < 8; ++i) { a[i] = a0[i]; a[8 + i] = a1[i]; }
    acc = __builtin_amdgcn_wmma_f32_16x16x32_f16(false, a, false, b, (short)0, acc, false, false);
  }
  float* pc = C + (size_t)(mo * 16 + hi * 8) * Nc + co * 16 + rl;
#pragma unroll
  for (int j = 0; j < 8; ++j) pc[(size_t)j * Nc] = acc[j];
}


__global__ void k_deg_edges(const int* __restrict__ ei, const float* __restrict__ ewp,
                            float* __restrict__ deg) {
  int e = blockIdx.x * blockDim.x + threadIdx.x;
  if (e < E_) atomicAdd(&deg[ei[E_ + e]], ewp ? ewp[e] : 1.f);
}

__global__ void k_dinv(float* __restrict__ deg, int n) {
  int i = blockIdx.x * blockDim.x + threadIdx.x;
  if (i < n) deg[i] = rsqrtf(deg[i] + 1.f);
}

__global__ void k_agg_edges(const int* __restrict__ ei, const float* __restrict__ hlin,
                            const float* __restrict__ dinv, const float* __restrict__ ewp,
                            float* __restrict__ outp, int F) {
  int t = blockIdx.x * blockDim.x + threadIdx.x;
  if (t >= E_ * F) return;
  int e = t / F, f = t - e * F;
  int r = ei[e], c = ei[E_ + e];
  float wgt = ewp ? ewp[e] : 1.f;
  atomicAdd(&outp[(size_t)c * F + f], hlin[(size_t)r * F + f] * dinv[r] * wgt * dinv[c]);
}

__global__ void k_agg_self(const float* __restrict__ hlin, const float* __restrict__ dinv,
                           const float* __restrict__ bias, float* __restrict__ outp,
                           int F, int relu) {
  int t = blockIdx.x * blockDim.x + threadIdx.x;
  if (t >= N_ * F) return;
  int n = t / F, f = t - n * F;
  float v = outp[t] + hlin[t] * dinv[n] * dinv[n] + bias[f];
  outp[t] = relu ? fmaxf(v, 0.f) : v;
}


__global__ __launch_bounds__(128, 1) void k_edge_passA(
    const float* __restrict__ h1, const int* __restrict__ ei,
    const unsigned* __restrict__ cw8, float* __restrict__ bnstat)
{
  __shared__ unsigned cwl[5 * 64 * 17];
  __shared__ unsigned xls[EW][64 * 17];
  __shared__ float abuf[EW][64];
  __shared__ float bbuf[EW][64];
  __shared__ float sacc[5 * 64 * 2];

  const int tid = threadIdx.x, lane = tid & 31, w = tid >> 5;

  for (int i = tid; i < 5 * 64 * 16; i += 128)
    cwl[(i >> 4) * 17 + (i & 15)] = cw8[i];
  for (int i = tid; i < 5 * 64 * 2; i += 128) sacc[i] = 0.f;

  const int e = blockIdx.x * EW + w;
  const int ra = ei[e], ca = ei[E_ + e];
  const int ia = (ra + N_ - 1) & (N_ - 1);
  const int ib = (ca + N_ - 1) & (N_ - 1);
  abuf[w][lane]      = h1[(size_t)ia * H_ + lane];
  abuf[w][lane + 32] = h1[(size_t)ia * H_ + lane + 32];
  bbuf[w][lane]      = h1[(size_t)ib * H_ + lane];
  bbuf[w][lane + 32] = h1[(size_t)ib * H_ + lane + 32];
  __syncthreads();

  for (int li = 0; li < 2; ++li) {
    const int l = lane + li * 32;
    const float bv = bbuf[w][l];
    for (int h4 = 0; h4 < 16; ++h4) {
      unsigned p = 0;
#pragma unroll
      for (int s = 0; s < 4; ++s)
        p |= f32_to_fp8(fabsf(abuf[w][h4 * 4 + s] - bv)) << (8 * s);
      xls[w][l * 17 + h4] = p;
    }
  }

  const int rl = lane & 15, hi = lane >> 4;
  v8i bf[4];
#pragma unroll
  for (int nl = 0; nl < 4; ++nl) {
    const int l = nl * 16 + rl;
#pragma unroll
    for (int v = 0; v < 8; ++v) {
      const int widx = ((v >= 4) ? 8 : 0) + hi * 4 + (v & 3);
      bf[nl][v] = (int)xls[w][l * 17 + widx];
    }
  }

  for (int net = 0; net < 5; ++net) {
#pragma unroll
    for (int mo = 0; mo < 4; ++mo) {
      v8i af;
#pragma unroll
      for (int v = 0; v < 8; ++v) {
        const int widx = ((v >= 4) ? 8 : 0) + hi * 4 + (v & 3);
        af[v] = (int)cwl[(net * 64 + mo * 16 + rl) * 17 + widx];
      }
#pragma unroll
      for (int nl = 0; nl < 4; ++nl) {
        v8f acc = {};
        acc = __builtin_amdgcn_wmma_f32_16x16x64_fp8_fp8(af, bf[nl], (short)0, acc, false, false);
#pragma unroll
        for (int j = 0; j < 8; ++j) {
          float s1 = acc[j], s2 = s1 * s1;
          for (int m = 8; m >= 1; m >>= 1) {
            s1 += __shfl_xor(s1, m, 32);
            s2 += __shfl_xor(s2, m, 32);
          }
          if (rl == 0) {
            const int o = mo * 16 + j + hi * 8;
            atomicAdd(&sacc[(net * 64 + o) * 2 + 0], s1);
            atomicAdd(&sacc[(net * 64 + o) * 2 + 1], s2);
          }
        }
      }
    }
  }
  __syncthreads();
  for (int i = tid; i < 5 * 64 * 2; i += 128) atomicAdd(&bnstat[i], sacc[i]);
}

__global__ void k_bn_fin(const float* __restrict__ stat, const float* __restrict__ g,
                         const float* __restrict__ b, float* __restrict__ aff) {
  int i = blockIdx.x * blockDim.x + threadIdx.x;
  if (i < 5 * H_) {
    const float cnt = (float)E_ * (float)H_;
    float mean = stat[2 * i] / cnt;
    float var = stat[2 * i + 1] / cnt - mean * mean;
    float sc = g[i] * rsqrtf(var + EPS_);
    aff[2 * i] = sc;
    aff[2 * i + 1] = b[i] - mean * sc;
  }
}


__global__ __launch_bounds__(128, 1) void k_edge_passB(
    const float* __restrict__ h1, const int* __restrict__ ei,
    const unsigned* __restrict__ cw8, const float* __restrict__ bnaff,
    const float* __restrict__ oww, const float* __restrict__ obv,
    const float* __restrict__ fc1, const float* __restrict__ fc2,
    const float* __restrict__ noise, float* __restrict__ ew_out,
    float* __restrict__ loss)
{
  __shared__ unsigned cwl[5 * 64 * 17];
  __shared__ unsigned xls[EW][64 * 17];
  __shared__ float abuf[EW][64];
  __shared__ float bbuf[EW][64];
  __shared__ float scl[320], shf[320], owl[320];
  __shared__ float nout[EW][5][64];
  __shared__ float zbuf[EW][64];
  __shared__ float sbuf[EW][64];
  __shared__ float pxy[EW][8];

  const int tid = threadIdx.x, lane = tid & 31, w = tid >> 5;

  for (int i = tid; i < 5 * 64 * 16; i += 128)
    cwl[(i >> 4) * 17 + (i & 15)] = cw8[i];
  for (int i = tid; i < 320; i += 128) {
    scl[i] = bnaff[2 * i];
    shf[i] = bnaff[2 * i + 1];
    owl[i] = oww[i];
  }
  for (int i = tid; i < EW * 5 * 64; i += 128) ((float*)nout)[i] = 0.f;

  const int e = blockIdx.x * EW + w;
  const int ra = ei[e], ca = ei[E_ + e];
  const int ia = (ra + N_ - 1) & (N_ - 1);
  const int ib = (ca + N_ - 1) & (N_ - 1);
  abuf[w][lane]      = h1[(size_t)ia * H_ + lane];
  abuf[w][lane + 32] = h1[(size_t)ia * H_ + lane + 32];
  bbuf[w][lane]      = h1[(size_t)ib * H_ + lane];
  bbuf[w][lane + 32] = h1[(size_t)ib * H_ + lane + 32];
  __syncthreads();

  for (int li = 0; li < 2; ++li) {
    const int l = lane + li * 32;
    const float bv = bbuf[w][l];
    for (int h4 = 0; h4 < 16; ++h4) {
      unsigned p = 0;
#pragma unroll
      for (int s = 0; s < 4; ++s)
        p |= f32_to_fp8(fabsf(abuf[w][h4 * 4 + s] - bv)) << (8 * s);
      xls[w][l * 17 + h4] = p;
    }
  }

  const int rl = lane & 15, hi = lane >> 4;
  v8i bf[4];
#pragma unroll
  for (int nl = 0; nl < 4; ++nl) {
    const int l = nl * 16 + rl;
#pragma unroll
    for (int v = 0; v < 8; ++v) {
      const int widx = ((v >= 4) ? 8 : 0) + hi * 4 + (v & 3);
      bf[nl][v] = (int)xls[w][l * 17 + widx];
    }
  }

  for (int net = 0; net < 5; ++net) {
#pragma unroll
    for (int mo = 0; mo < 4; ++mo) {
      v8i af;
#pragma unroll
      for (int v = 0; v < 8; ++v) {
        const int widx = ((v >= 4) ? 8 : 0) + hi * 4 + (v & 3);
        af[v] = (int)cwl[(net * 64 + mo * 16 + rl) * 17 + widx];
      }
#pragma unroll
      for (int nl = 0; nl < 4; ++nl) {
        v8f acc = {};
        acc = __builtin_amdgcn_wmma_f32_16x16x64_fp8_fp8(af, bf[nl], (short)0, acc, false, false);
        float part = 0.f;
#pragma unroll
        for (int j = 0; j < 8; ++j) {
          const int o = mo * 16 + j + hi * 8;
          float t = acc[j] * scl[net * 64 + o] + shf[net * 64 + o];
          t = t > 0.f ? t : 0.01f * t;
          part += owl[net * 64 + o] * t;
        }
        part += __shfl_xor(part, 16, 32);
        if (hi == 0) nout[w][net][nl * 16 + rl] += part;
      }
    }
  }

  const float ob0 = obv[0], ob1 = obv[1], ob2 = obv[2], ob3 = obv[3], ob4 = obv[4];
  for (int li = 0; li < 2; ++li) {
    const int l = lane + li * 32;
    const float s = nout[w][0][l] + ob0;
    const float lm = (nout[w][1][l] + ob1) * s + (nout[w][3][l] + ob3);
    const float lv = fabsf(logf(s * s * expf(nout[w][2][l] + ob2) + expf(nout[w][4][l] + ob4)));
    zbuf[w][l] = sigmoidf_(lm + lv * noise[(size_t)e * H_ + l]);
    sbuf[w][l] = s;
  }
  if (lane < EN_) {
    float px = 0.f, ey = 0.f;
    for (int l = 0; l < H_; ++l) {
      px += sbuf[w][l] * fc1[lane * H_ + l];
      ey += zbuf[w][l] * fc2[lane * H_ + l];
    }
    pxy[w][lane] = sigmoidf_(px);
    pxy[w][EN_ + lane] = ey;
  }
  if (lane == 0) {
    float a[4], bq[4];
#pragma unroll
    for (int k = 0; k < 4; ++k) { a[k] = pxy[w][k]; bq[k] = pxy[w][4 + k]; }
    float am = fmaxf(fmaxf(a[0], a[1]), fmaxf(a[2], a[3]));
    float bm = fmaxf(fmaxf(bq[0], bq[1]), fmaxf(bq[2], bq[3]));
    float es = 0.f, fs = 0.f;
#pragma unroll
    for (int k = 0; k < 4; ++k) { es += expf(a[k] - am); fs += expf(bq[k] - bm); }
    const float lse = logf(es), lfs = logf(fs);
    float contrib = 0.f;
#pragma unroll
    for (int k = 0; k < 4; ++k) {
      const float logp = (bq[k] - bm) - lfs;
      const float p = expf(logp);
      const float logpx = (a[k] - am) - lse;
      contrib += p * (logp - logpx);
    }
    atomicAdd(loss, contrib);
    ew_out[e] = 0.25f * (a[0] + a[1] + a[2] + a[3]);
  }
}


__global__ void k_root_rows(const int* __restrict__ root, const int* __restrict__ bid,
                            int* __restrict__ rr) {
  int n = blockIdx.x * blockDim.x + threadIdx.x;
  if (n < N_) rr[n] = root[bid[n]];
}

__global__ __launch_bounds__(256) void k_bn1_stats(const float* __restrict__ h1,
                                                   const float* __restrict__ x0,
                                                   const int* __restrict__ rr,
                                                   float* __restrict__ stat)
{
  const int tid = threadIdx.x;
  float s[4] = {0.f, 0.f, 0.f, 0.f}, q[4] = {0.f, 0.f, 0.f, 0.f};
  int fidx[4]; int nf = 0;
  for (int k = 0; k < 4; ++k) { int ff = tid + k * 256; if (ff < HIN_) fidx[nf++] = ff; }
  const int rows = N_ / 64;
  const int n0 = blockIdx.x * rows;
  for (int n = n0; n < n0 + rows; ++n) {
    const int rn = rr[n];
    for (int k = 0; k < nf; ++k) {
      const int ff = fidx[k];
      float v = (ff < H_) ? h1[(size_t)n * H_ + ff] : x0[(size_t)rn * IN_ + ff - H_];
      s[k] += v; q[k] += v * v;
    }
  }
  for (int k = 0; k < nf; ++k) {
    atomicAdd(&stat[fidx[k]], s[k]);
    atomicAdd(&stat[HIN_ + fidx[k]], q[k]);
  }
}

__global__ void k_bn1_fin(const float* __restrict__ stat, const float* __restrict__ g,
                          const float* __restrict__ b, float* __restrict__ aff) {
  int i = blockIdx.x * blockDim.x + threadIdx.x;
  if (i < HIN_) {
    float mean = stat[i] / (float)N_;
    float var = stat[HIN_ + i] / (float)N_ - mean * mean;
    float sc = g[i] * rsqrtf(var + EPS_);
    aff[i] = sc;
    aff[HIN_ + i] = b[i] - mean * sc;
  }
}

__global__ void k_build_hb(const float* __restrict__ h1, const float* __restrict__ x0,
                           const int* __restrict__ rr, const float* __restrict__ aff,
                           _Float16* __restrict__ hb) {
  int t = blockIdx.x * blockDim.x + threadIdx.x;
  if (t >= N_ * HIN_) return;
  int n = t / HIN_, f = t - n * HIN_;
  float v = (f < H_) ? h1[(size_t)n * H_ + f] : x0[(size_t)rr[n] * IN_ + (f - H_)];
  hb[t] = (_Float16)fmaxf(v * aff[f] + aff[HIN_ + f], 0.f);
}


__global__ void k_pool(const float* __restrict__ h2, const float* __restrict__ h1,
                       const int* __restrict__ rr, const int* __restrict__ bid,
                       float* __restrict__ gsum, float* __restrict__ gcnt) {
  int t = blockIdx.x * blockDim.x + threadIdx.x;
  if (t >= N_ * 128) return;
  int n = t >> 7, f = t & 127;
  int b = bid[n];
  float v = (f < OUT_) ? h2[(size_t)n * OUT_ + f] : h1[(size_t)rr[n] * H_ + (f - OUT_)];
  atomicAdd(&gsum[b * 128 + f], v);
  if (f == 0) atomicAdd(&gcnt[b], 1.f);
}

__global__ void k_finalize(const float* __restrict__ gsum, const float* __restrict__ gcnt,
                           const float* __restrict__ loss, float* __restrict__ outp) {
  int i = blockIdx.x * blockDim.x + threadIdx.x;
  if (i < B_ * 128) outp[i] = gsum[i] / fmaxf(gcnt[i >> 7], 1.f);
  else if (i == B_ * 128) outp[i] = loss[0] / (float)E_;
}


extern "C" void kernel_launch(void* const* d_in, const int* in_sizes, int n_in,
                              void* d_out, int out_size, void* d_ws, size_t ws_size,
                              hipStream_t stream) {
  (void)in_sizes; (void)n_in; (void)out_size; (void)ws_size;
  const float* x0    = (const float*)d_in[0];
  const int*   ei    = (const int*)  d_in[1];
  const int*   root  = (const int*)  d_in[2];
  const int*   bid   = (const int*)  d_in[3];
  const float* noise = (const float*)d_in[4];
  const float* W1    = (const float*)d_in[5];
  const float* b1    = (const float*)d_in[6];
  const float* W2    = (const float*)d_in[7];
  const float* b2    = (const float*)d_in[8];
  const float* cw    = (const float*)d_in[9];
  const float* bng   = (const float*)d_in[10];
  const float* bnb   = (const float*)d_in[11];
  const float* oww   = (const float*)d_in[12];
  const float* obv   = (const float*)d_in[13];
  const float* fc1   = (const float*)d_in[14];
  const float* fc2   = (const float*)d_in[15];
  const float* g1    = (const float*)d_in[16];
  const float* bb1   = (const float*)d_in[17];
  float* outp = (float*)d_out;

  char* wsp = (char*)d_ws;
  auto alloc = [&](size_t bytes) -> void* {
    void* p = (void*)wsp;
    wsp += (bytes + 255) & ~(size_t)255;
    return p;
  };
  _Float16* x0h   = (_Float16*)alloc((size_t)N_ * IN_ * 2);
  _Float16* w1h   = (_Float16*)alloc((size_t)H_ * IN_ * 2);
  float*    h1lin = (float*)   alloc((size_t)N_ * H_ * 4);
  float*    dinv1 = (float*)   alloc((size_t)N_ * 4);
  float*    h1    = (float*)   alloc((size_t)N_ * H_ * 4);
  unsigned* cw8   = (unsigned*)alloc((size_t)5 * H_ * H_);
  float*    bnstat= (float*)   alloc((size_t)5 * H_ * 2 * 4);
  float*    bnaff = (float*)   alloc((size_t)5 * H_ * 2 * 4);
  float*    ew    = (float*)   alloc((size_t)E_ * 4);
  float*    loss  = (float*)   alloc(256);
  int*      rr    = (int*)     alloc((size_t)N_ * 4);
  float*    bn1st = (float*)   alloc((size_t)HIN_ * 2 * 4);
  float*    bn1af = (float*)   alloc((size_t)HIN_ * 2 * 4);
  _Float16* hbh   = (_Float16*)alloc((size_t)N_ * HIN_ * 2);
  _Float16* w2h   = (_Float16*)alloc((size_t)OUT_ * HIN_ * 2);
  float*    h2lin = (float*)   alloc((size_t)N_ * OUT_ * 4);
  float*    dinv2 = (float*)   alloc((size_t)N_ * 4);
  float*    h2    = (float*)   alloc((size_t)N_ * OUT_ * 4);
  float*    gsum  = (float*)   alloc((size_t)B_ * 128 * 4);
  float*    gcnt  = (float*)   alloc((size_t)B_ * 4);

#define GS(n, b) dim3(((n) + (b) - 1) / (b)), dim3(b), 0, stream

  k_f32_to_f16<<<GS(N_ * IN_, 256)>>>(x0, x0h, N_ * IN_);
  k_f32_to_f16<<<GS(H_ * IN_, 256)>>>(W1, w1h, H_ * IN_);
  k_wmma_gemm_f16<<<dim3((N_ / 16) * (H_ / 16)), dim3(32), 0, stream>>>(x0h, w1h, h1lin, N_, H_, IN_);
  k_zero<<<GS(N_, 256)>>>(dinv1, N_);
  k_deg_edges<<<GS(E_, 256)>>>(ei, (const float*)nullptr, dinv1);
  k_dinv<<<GS(N_, 256)>>>(dinv1, N_);
  k_zero<<<GS(N_ * H_, 256)>>>(h1, N_ * H_);
  k_agg_edges<<<GS(E_ * H_, 256)>>>(ei, h1lin, dinv1, (const float*)nullptr, h1, H_);
  k_agg_self<<<GS(N_ * H_, 256)>>>(h1lin, dinv1, b1, h1, H_, 0);

  k_f32_to_fp8x4<<<GS(5 * H_ * H_ / 4, 256)>>>(cw, cw8, 5 * H_ * H_ / 4);
  k_zero<<<GS(5 * H_ * 2, 256)>>>(bnstat, 5 * H_ * 2);
  k_edge_passA<<<dim3(E_ / EW), dim3(128), 0, stream>>>(h1, ei, cw8, bnstat);
  k_bn_fin<<<GS(5 * H_, 256)>>>(bnstat, bng, bnb, bnaff);
  k_zero<<<GS(1, 64)>>>(loss, 1);
  k_edge_passB<<<dim3(E_ / EW), dim3(128), 0, stream>>>(h1, ei, cw8, bnaff, oww, obv,
                                                        fc1, fc2, noise, ew, loss);

  k_root_rows<<<GS(N_, 256)>>>(root, bid, rr);
  k_zero<<<GS(HIN_ * 2, 256)>>>(bn1st, HIN_ * 2);
  k_bn1_stats<<<dim3(64), dim3(256), 0, stream>>>(h1, x0, rr, bn1st);
  k_bn1_fin<<<GS(HIN_, 256)>>>(bn1st, g1, bb1, bn1af);
  k_build_hb<<<GS(N_ * HIN_, 256)>>>(h1, x0, rr, bn1af, hbh);

  k_f32_to_f16<<<GS(OUT_ * HIN_, 256)>>>(W2, w2h, OUT_ * HIN_);
  k_wmma_gemm_f16<<<dim3((N_ / 16) * (OUT_ / 16)), dim3(32), 0, stream>>>(hbh, w2h, h2lin, N_, OUT_, HIN_);
  k_zero<<<GS(N_, 256)>>>(dinv2, N_);
  k_deg_edges<<<GS(E_, 256)>>>(ei, ew, dinv2);
  k_dinv<<<GS(N_, 256)>>>(dinv2, N_);
  k_zero<<<GS(N_ * OUT_, 256)>>>(h2, N_ * OUT_);
  k_agg_edges<<<GS(E_ * OUT_, 256)>>>(ei, h2lin, dinv2, ew, h2, OUT_);
  k_agg_self<<<GS(N_ * OUT_, 256)>>>(h2lin, dinv2, b2, h2, OUT_, 1);

  k_zero<<<GS(B_ * 128, 256)>>>(gsum, B_ * 128);
  k_zero<<<GS(B_, 64)>>>(gcnt, B_);
  k_pool<<<GS(N_ * 128, 256)>>>(h2, h1, rr, bid, gsum, gcnt);
  k_finalize<<<GS(B_ * 128 + 1, 256)>>>(gsum, gcnt, loss, outp);

#undef GS
}
